// SNNCore_56985626083433
// MI455X (gfx1250) — matrix operand order repaired and checked by reading, not compiled, not run on hardware
//
#include <hip/hip_runtime.h>

// ---------------------------------------------------------------------------
// CDNA5 (gfx1250, wave32) implementation of the SNN predictive-coding model.
//   K0: one-shot f32->bf16 weight conversion (enc_W, gen_W, inf_W, out_W)
//   K1: embedding gather + encoder GEMM      (bf16 WMMA, f32 accum)
//   K2: persistent single-WGP recurrent scan (bf16 WMMA, f32 accum, LDS state)
//       512 threads (16 waves, 4/SIMD) for VGPR headroom; weight fragments
//       stream through a 2-deep pipeline of 4-Kstep batches (64 VGPRs peak)
//       so the sequential chain exposes ~1 L2 latency per matmul, no spills
//   K3: vocab projection GEMM 2048x32000x1024 (bf16 WMMA, f32 accum) -- the
//       134 GFLOP hot spot, compute-bound. B operands pre-converted to bf16
//       (64 MB, L2-resident) and staged with GLOBAL_LOAD_ASYNC_TO_LDS_B128
//       in a double-buffered ASYNCcnt pipeline.
// ---------------------------------------------------------------------------

typedef __attribute__((ext_vector_type(16))) __bf16 v16bf;
typedef __attribute__((ext_vector_type(8)))  float  v8f;

union FragCast { uint4 q[2]; v16bf v; };

__device__ __forceinline__ v16bf make_frag(uint4 lo, uint4 hi) {
  FragCast u; u.q[0] = lo; u.q[1] = hi; return u.v;
}

__device__ __forceinline__ v8f wmma_bf16(v16bf a, v16bf b, v8f c) {
  // v_wmma_f32_16x16x32_bf16  (D = A x B + C, f32 accumulate)
  return __builtin_amdgcn_wmma_f32_16x16x32_bf16(false, a, false, b,
                                                 (short)0, c, false, false);
}

// round-to-nearest-even f32 -> bf16
__device__ __forceinline__ unsigned short bf16b(float f) {
  union { float f; unsigned int u; } x; x.f = f;
  unsigned int r = x.u + 0x7FFFu + ((x.u >> 16) & 1u);
  return (unsigned short)(r >> 16);
}
__device__ __forceinline__ unsigned int pk_bf16(float a, float b) {
  return (unsigned int)bf16b(a) | ((unsigned int)bf16b(b) << 16);
}

// ---- CDNA5 async global->LDS copy (ASYNCcnt-tracked, no VGPR round trip).
// IOFFSET is added to BOTH the LDS and the global address (ISA 10. / 15.18.3),
// so one (lds,vaddr) pair covers two consecutive b128 transfers.
__device__ __forceinline__ void async_copy_32B(void* ldsDst, const void* base,
                                               unsigned int byteOff) {
  unsigned int ldsAddr = (unsigned int)(size_t)ldsDst;  // LDS aperture low bits
  asm volatile(
      "global_load_async_to_lds_b128 %0, %1, %2\n\t"
      "global_load_async_to_lds_b128 %0, %1, %2 offset:16"
      :
      : "v"(ldsAddr), "v"(byteOff), "s"(base)
      : "memory");
}
__device__ __forceinline__ void wait_async0() {
  asm volatile("s_wait_asynccnt 0x0" ::: "memory");
}

// ---------------------------------------------------------------------------
// K0: bulk f32 -> bf16 conversion (n must be a multiple of 8; all our sizes are)
// ---------------------------------------------------------------------------
__global__ __launch_bounds__(256) void cvt_bf16(const float* __restrict__ src,
                                                unsigned short* __restrict__ dst,
                                                int n) {
  int i = (blockIdx.x * 256 + threadIdx.x) * 8;
  if (i < n) {
    const float4* s = (const float4*)(src + i);
    float4 a = s[0], b = s[1];
    uint4 o = make_uint4(pk_bf16(a.x, a.y), pk_bf16(a.z, a.w),
                         pk_bf16(b.x, b.y), pk_bf16(b.z, b.w));
    *(uint4*)(dst + i) = o;
  }
}

// ---------------------------------------------------------------------------
// Generic 128x128-tile WMMA GEMM:  Out[m,n] = sum_k A[m,k]*Bbf[n,k] + bias[n]
//   MODE 0: A rows gathered from f32 table (emb[ids[m]]), converted to bf16
//   MODE 1: A rows already bf16 (hidden states) -> async LDS staging
// B is always bf16 (pre-converted) -> async LDS staging, double buffered.
// 256 threads = 8 waves; wave grid 4(M) x 2(N); wave tile 32x64 = 2x4 WMMAs.
// LDS tiles use 20-uint (80B) row pitch -> conflict-free ds_load_b128 frags.
// ---------------------------------------------------------------------------
template <int MODE>
__global__ __launch_bounds__(256) void wmma_gemm(
    const float* __restrict__ Af32, const unsigned short* __restrict__ Abf,
    const int* __restrict__ ids, const unsigned short* __restrict__ Bbf,
    const float* __restrict__ bias, float* __restrict__ Out,
    int Mtot, int Ntot, int K) {
  __shared__ unsigned int sA[2][128 * 20];
  __shared__ unsigned int sB[2][128 * 20];

  const int tid  = threadIdx.x;
  const int lane = tid & 31;
  const int wid  = tid >> 5;
  const int rl   = lane & 15;
  const int half = lane >> 4;
  const int waveM = wid & 3;
  const int waveN = wid >> 2;
  const int M0 = blockIdx.y * 128;
  const int N0 = blockIdx.x * 128;

  const int r_ld    = tid >> 1;        // 0..127: tile row this thread stages
  const int c_ld    = (tid & 1) * 16;  // element offset within 32-wide K slab
  const int lds_off = r_ld * 20 + (tid & 1) * 8;

  int gatherRow = 0;
  if (MODE == 0) gatherRow = ids[M0 + r_ld];

  auto stage = [&](int buf, int k0v) {
    if constexpr (MODE == 0) {
      // gather + in-register convert (encoder only; tiny kernel)
      const float4* src =
          (const float4*)(Af32 + (size_t)gatherRow * K + k0v + c_ld);
      float4 f0 = src[0], f1 = src[1], f2 = src[2], f3 = src[3];
      uint4 lo = make_uint4(pk_bf16(f0.x, f0.y), pk_bf16(f0.z, f0.w),
                            pk_bf16(f1.x, f1.y), pk_bf16(f1.z, f1.w));
      uint4 hi = make_uint4(pk_bf16(f2.x, f2.y), pk_bf16(f2.z, f2.w),
                            pk_bf16(f3.x, f3.y), pk_bf16(f3.z, f3.w));
      *(uint4*)&sA[buf][lds_off]     = lo;
      *(uint4*)&sA[buf][lds_off + 4] = hi;
    } else {
      async_copy_32B(&sA[buf][lds_off], Abf,
                     (unsigned int)(((size_t)(M0 + r_ld) * K + k0v + c_ld) * 2));
    }
    async_copy_32B(&sB[buf][lds_off], Bbf,
                   (unsigned int)(((size_t)(N0 + r_ld) * K + k0v + c_ld) * 2));
  };

  const v8f vzero = {0.f, 0.f, 0.f, 0.f, 0.f, 0.f, 0.f, 0.f};
  v8f acc[2][4];
#pragma unroll
  for (int m = 0; m < 2; ++m)
#pragma unroll
    for (int n = 0; n < 4; ++n) acc[m][n] = vzero;

  stage(0, 0);
  wait_async0();
  __syncthreads();

  int p = 0;
  for (int k0 = 0; k0 < K; k0 += 32) {
    if (k0 + 32 < K) stage(p ^ 1, k0 + 32);  // prefetch next slab (async)

    // ---- B fragments: lane <-> N column, 16 contiguous K per lane-half
    v16bf bfr[4];
#pragma unroll
    for (int n = 0; n < 4; ++n) {
      const unsigned int* q = &sB[p][(waveN * 64 + n * 16 + rl) * 20 + half * 8];
      bfr[n] = make_frag(*(const uint4*)q, *(const uint4*)(q + 4));
    }
    // ---- A fragments: lane <-> M row, 16 contiguous K per lane-half
#pragma unroll
    for (int m = 0; m < 2; ++m) {
      const unsigned int* q = &sA[p][(waveM * 32 + m * 16 + rl) * 20];
      v16bf af = make_frag(*(const uint4*)(q + half * 8),
                           *(const uint4*)(q + 4 + half * 8));
#pragma unroll
      for (int n = 0; n < 4; ++n) acc[m][n] = wmma_bf16(af, bfr[n], acc[m][n]);
    }

    wait_async0();     // our async writes to buffer p^1 have landed
    __syncthreads();   // everyone's have
    p ^= 1;
  }

  // ---- epilogue: D layout -> VGPR r, lanes0-15:M=r, lanes16-31:M=r+8, N=lane
#pragma unroll
  for (int n = 0; n < 4; ++n) {
    const int gn = N0 + waveN * 64 + n * 16 + rl;
    const float bb = bias[gn];
#pragma unroll
    for (int m = 0; m < 2; ++m) {
      const int gm = M0 + waveM * 32 + m * 16 + half * 8;
#pragma unroll
      for (int r = 0; r < 8; ++r)
        Out[(size_t)(gm + r) * Ntot + gn] = acc[m][n][r] + bb;
    }
  }
}

// ---------------------------------------------------------------------------
// Persistent recurrent scan: one 512-thread workgroup (16 wave32s, 4/SIMD for
// VGPR headroom -> no scratch spills on the critical path).
// States/errors live in LDS; normalized activations staged as bf16 with a
// 16B-granular XOR swizzle so wmma A-fragment ds_load_b128s are conflict-free.
// Weights arrive pre-converted bf16 (L2-resident across all 512 steps) and
// stream through a 2-deep pipeline of 4-Kstep fragment batches: batch b+1 is
// issued while batch b feeds the WMMAs, bounding live weight regs to 64.
// M=16 WMMA tiles are zero-padded past batch B=8.
// ---------------------------------------------------------------------------
__global__ __launch_bounds__(512) void scan_kernel(
    const float* __restrict__ H0, const unsigned short* __restrict__ genWb,
    const float* __restrict__ genB, const unsigned short* __restrict__ infWb,
    const float* __restrict__ infB, const float* __restrict__ g_ns,
    const float* __restrict__ b_ns, const float* __restrict__ g_ne,
    const float* __restrict__ b_ne, const float* __restrict__ errS,
    const int* __restrict__ tsteps, unsigned short* __restrict__ hid) {
  __shared__ float st[4][8][256];           // persistent layer states (32 KB)
  __shared__ float bu[8][512];              // bottom-up / error       (16 KB)
  __shared__ unsigned short aln[16 * 512];  // bf16 WMMA A operand     (16 KB)

  const int tid  = threadIdx.x;
  const int lane = tid & 31;
  const int wid  = tid >> 5;  // 0..15
  const int rl   = lane & 15;
  const int half = lane >> 4;
  const v8f vzero = {0.f, 0.f, 0.f, 0.f, 0.f, 0.f, 0.f, 0.f};

  for (int i = tid; i < 4 * 8 * 256; i += 512) (&st[0][0][0])[i] = 0.f;
  for (int i = tid; i < 16 * 512; i += 512) aln[i] = 0;  // rows 8..15 stay 0
  __syncthreads();

  const int T = tsteps[0];
  for (int t = 0; t < T; ++t) {
    for (int s = 0; s < 256; ++s) {
      // x_t = H0[:, s, :]
      for (int i = tid; i < 8 * 512; i += 512) {
        int b = i >> 9, d = i & 511;
        bu[b][d] = H0[((size_t)b * 256 + s) * 512 + d];
      }
      __syncthreads();

      for (int j = 0; j < 4; ++j) {
        // ---- LayerNorm(state) -> aln, K=256 (one wave per batch row)
        if (wid < 8) {
          float xv[8], s1 = 0.f, s2 = 0.f;
#pragma unroll
          for (int i = 0; i < 8; ++i) {
            float x = st[j][wid][i * 32 + lane];
            xv[i] = x; s1 += x; s2 += x * x;
          }
#pragma unroll
          for (int o = 16; o > 0; o >>= 1) {
            s1 += __shfl_xor(s1, o);
            s2 += __shfl_xor(s2, o);
          }
          float mean = s1 * (1.f / 256.f);
          float var  = s2 * (1.f / 256.f) - mean * mean;
          float rs   = rsqrtf(var + 1e-5f);
#pragma unroll
          for (int i = 0; i < 8; ++i) {
            int idx = i * 32 + lane;
            float y = (xv[i] - mean) * rs * g_ns[j * 256 + idx] + b_ns[j * 256 + idx];
            int phys = (idx >> 3) ^ wid;  // 16B-chunk XOR swizzle
            aln[wid * 512 + phys * 8 + (idx & 7)] = bf16b(y);
          }
        }
        __syncthreads();

        // ---- generative matmul: pred = LN(st) @ gen_W[j]^T  (N=512, K=256)
        // 32 N-tiles over 16 waves: each wave runs tiles wid and wid+16.
        {
          const unsigned int* A = (const unsigned int*)aln;
          const float es = errS[j];
#pragma unroll
          for (int tt = 0; tt < 2; ++tt) {
            const int n0 = (wid + 16 * tt) * 16;
            const unsigned short* wrow =
                genWb + (size_t)j * 512 * 256 + (size_t)(n0 + rl) * 256 + half * 16;
            v8f acc = vzero;
            uint4 wf[2][4][2];  // two 4-Kstep batches in flight (64 VGPRs peak)
#pragma unroll
            for (int i = 0; i < 4; ++i) {
              const uint4* wp = (const uint4*)(wrow + i * 32);
              wf[0][i][0] = wp[0]; wf[0][i][1] = wp[1];
            }
#pragma unroll
            for (int b = 0; b < 2; ++b) {
              if (b + 1 < 2) {
#pragma unroll
                for (int i = 0; i < 4; ++i) {
                  const uint4* wp = (const uint4*)(wrow + (b + 1) * 128 + i * 32);
                  wf[b + 1][i][0] = wp[0]; wf[b + 1][i][1] = wp[1];
                }
              }
#pragma unroll
              for (int i = 0; i < 4; ++i) {
                int kk = b * 128 + i * 32;
                int c0 = (kk >> 3) + 2 * half;
                uint4 alo = *(const uint4*)&A[rl * 256 + ((c0 ^ rl) << 2)];
                uint4 ahi = *(const uint4*)&A[rl * 256 + (((c0 + 1) ^ rl) << 2)];
                acc = wmma_bf16(make_frag(alo, ahi),
                                make_frag(wf[b][i][0], wf[b][i][1]), acc);
              }
            }
            if (half == 0) {  // lanes 0-15 hold M=0..7 (valid batch rows)
              const int n = n0 + lane;
              const float gb = genB[j * 512 + n];
#pragma unroll
              for (int r = 0; r < 8; ++r) {
                float pr = (acc[r] + gb) > 1.0f ? 1.0f : 0.0f;  // spike
                bu[r][n] = (bu[r][n] - pr) * es;                // error
              }
            }
          }
        }
        __syncthreads();

        // ---- LayerNorm(error) -> aln, K=512
        if (wid < 8) {
          float xv[16], s1 = 0.f, s2 = 0.f;
#pragma unroll
          for (int i = 0; i < 16; ++i) {
            float x = bu[wid][i * 32 + lane];
            xv[i] = x; s1 += x; s2 += x * x;
          }
#pragma unroll
          for (int o = 16; o > 0; o >>= 1) {
            s1 += __shfl_xor(s1, o);
            s2 += __shfl_xor(s2, o);
          }
          float mean = s1 * (1.f / 512.f);
          float var  = s2 * (1.f / 512.f) - mean * mean;
          float rs   = rsqrtf(var + 1e-5f);
#pragma unroll
          for (int i = 0; i < 16; ++i) {
            int idx = i * 32 + lane;
            float y = (xv[i] - mean) * rs * g_ne[j * 512 + idx] + b_ne[j * 512 + idx];
            int phys = (idx >> 3) ^ wid;
            aln[wid * 512 + phys * 8 + (idx & 7)] = bf16b(y);
          }
        }
        __syncthreads();

        // ---- inference matmul: upd = LN(err) @ inf_W[j]^T  (N=256, K=512)
        // 16 N-tiles over 16 waves; 4 batches of 4 Ksteps, 2-deep pipeline.
        {
          const int n0 = wid * 16;
          const unsigned short* wrow =
              infWb + (size_t)j * 256 * 512 + (size_t)(n0 + rl) * 512 + half * 16;
          v8f acc = vzero;
          const unsigned int* A = (const unsigned int*)aln;
          uint4 wf[2][4][2];
#pragma unroll
          for (int i = 0; i < 4; ++i) {
            const uint4* wp = (const uint4*)(wrow + i * 32);
            wf[0][i][0] = wp[0]; wf[0][i][1] = wp[1];
          }
#pragma unroll
          for (int b = 0; b < 4; ++b) {
            const int cur = b & 1;
            if (b + 1 < 4) {
              const int nxt = cur ^ 1;
#pragma unroll
              for (int i = 0; i < 4; ++i) {
                const uint4* wp = (const uint4*)(wrow + (b + 1) * 128 + i * 32);
                wf[nxt][i][0] = wp[0]; wf[nxt][i][1] = wp[1];
              }
            }
#pragma unroll
            for (int i = 0; i < 4; ++i) {
              int kk = b * 128 + i * 32;
              int c0 = (kk >> 3) + 2 * half;
              uint4 alo = *(const uint4*)&A[rl * 256 + ((c0 ^ rl) << 2)];
              uint4 ahi = *(const uint4*)&A[rl * 256 + (((c0 + 1) ^ rl) << 2)];
              acc = wmma_bf16(make_frag(alo, ahi),
                              make_frag(wf[cur][i][0], wf[cur][i][1]), acc);
            }
          }
          if (half == 0) {
            const int n = n0 + lane;
            const float ib = infB[j * 256 + n];
#pragma unroll
            for (int r = 0; r < 8; ++r) {
              float u = (acc[r] + ib) > 1.0f ? 1.0f : 0.0f;  // spike
              float v = st[j][r][n] * 0.9f + u * 0.1f;       // leaky update
              st[j][r][n] = v;
              // hidden[b, s, j*DS + n] as bf16 for the vocab GEMM
              hid[((size_t)r * 256 + s) * 1024 + j * 256 + n] = bf16b(v);
            }
          }
        }
        __syncthreads();
      }  // layers
    }    // tokens
  }      // time steps
}

// ---------------------------------------------------------------------------
extern "C" void kernel_launch(void* const* d_in, const int* in_sizes, int n_in,
                              void* d_out, int out_size, void* d_ws,
                              size_t ws_size, hipStream_t stream) {
  const int*   ids  = (const int*)d_in[0];
  const float* emb  = (const float*)d_in[1];
  const float* encW = (const float*)d_in[2];
  const float* encB = (const float*)d_in[3];
  const float* genW = (const float*)d_in[4];
  const float* genB = (const float*)d_in[5];
  const float* infW = (const float*)d_in[6];
  const float* infB = (const float*)d_in[7];
  const float* nsg  = (const float*)d_in[8];
  const float* nsb  = (const float*)d_in[9];
  const float* neg2 = (const float*)d_in[10];
  const float* neb2 = (const float*)d_in[11];
  const float* errs = (const float*)d_in[12];
  const float* outW = (const float*)d_in[13];
  const float* outB = (const float*)d_in[14];
  const int*   tst  = (const int*)d_in[15];

  // workspace layout (bytes):
  //   H0   f32  [2048,512]      @ 0         (4 MB)
  //   hid  bf16 [2048,1024]     @ 4 MB      (4 MB)
  //   encWb bf16 [512,512]      @ 8 MB      (0.5 MB)
  //   genWb bf16 [4,512,256]    @ 8.5 MB    (1 MB)
  //   infWb bf16 [4,256,512]    @ 9.5 MB    (1 MB)
  //   outWb bf16 [32000,1024]   @ 10.5 MB   (62.5 MB)
  char* ws = (char*)d_ws;
  float* H0 = (float*)ws;
  unsigned short* hid   = (unsigned short*)(ws + (size_t)4 * 1024 * 1024);
  unsigned short* encWb = (unsigned short*)(ws + (size_t)8 * 1024 * 1024);
  unsigned short* genWb = (unsigned short*)(ws + (size_t)8704 * 1024);
  unsigned short* infWb = (unsigned short*)(ws + (size_t)9728 * 1024);
  unsigned short* outWb = (unsigned short*)(ws + (size_t)10752 * 1024);
  float* out = (float*)d_out;

  // K0: one-shot weight conversions (8 elems/thread, 2048 elems/block)
  cvt_bf16<<<128,   256, 0, stream>>>(encW, encWb, 512 * 512);
  cvt_bf16<<<256,   256, 0, stream>>>(genW, genWb, 4 * 512 * 256);
  cvt_bf16<<<256,   256, 0, stream>>>(infW, infWb, 4 * 256 * 512);
  cvt_bf16<<<16000, 256, 0, stream>>>(outW, outWb, 32000 * 1024);

  // K1: encoder  M=2048 N=512 K=512
  wmma_gemm<0><<<dim3(4, 16), 256, 0, stream>>>(emb, nullptr, ids, encWb, encB,
                                                H0, 2048, 512, 512);
  // K2: sequential scan (one WGP, 16 waves)
  scan_kernel<<<1, 512, 0, stream>>>(H0, genWb, genB, infWb, infB, nsg, nsb,
                                     neg2, neb2, errs, tst, hid);
  // K3: vocab projection  M=2048 N=32000 K=1024  (dominant, compute-bound)
  wmma_gemm<1><<<dim3(250, 16), 256, 0, stream>>>(nullptr, hid, nullptr, outWb,
                                                  outB, out, 2048, 32000, 1024);
}
